// HierarchicalResidualQuantizer_26448408609285
// MI455X (gfx1250) — matrix operand order repaired and checked by reading, not compiled, not run on hardware
//
#include <hip/hip_runtime.h>
#include <cstdint>
#include <cstddef>

// ---------------- CDNA5 WMMA types ----------------
typedef __attribute__((ext_vector_type(16))) __bf16 v16bf;
typedef __attribute__((ext_vector_type(8)))  float  v8f;
typedef unsigned int u32x4 __attribute__((ext_vector_type(4)));
typedef unsigned int u32x2 __attribute__((ext_vector_type(2)));
typedef float        f32x4 __attribute__((ext_vector_type(4)));

#define HRQ_L 8
#define HRQ_K 512
#define HRQ_D 256
#define HRQ_N 16384     // 16*32*32 tokens
#define TOK_WG 64       // tokens per workgroup
#define KBLK   64       // codes per k-block
#define NKB    (HRQ_K / KBLK)            // 8 k-blocks
#define RSTR   264      // padded ushort row stride for bf16 tiles (16B-aligned rows, conflict-free)
#define SSTR   65       // padded float row stride for score sub-tile
#define LOGK   6.2383246250395075f       // log(512)

struct F8frag { u32x4 lo; u32x4 hi; };
struct HiLo   { unsigned int hi, lo; };

__device__ __forceinline__ unsigned short f2bf(float f) {
  unsigned int u = __float_as_uint(f);
  u += 0x7FFFu + ((u >> 16) & 1u);          // round-to-nearest-even
  return (unsigned short)(u >> 16);
}
__device__ __forceinline__ float bf2f(unsigned short h) {
  return __uint_as_float(((unsigned int)h) << 16);
}
// split a pair of floats into packed bf16 hi and compensation lo words
__device__ __forceinline__ HiLo split2(float a, float b) {
  HiLo r;
  unsigned short ha = f2bf(a), hb = f2bf(b);
  r.hi = (unsigned int)ha | ((unsigned int)hb << 16);
  unsigned short la = f2bf(a - bf2f(ha)), lb = f2bf(b - bf2f(hb));
  r.lo = (unsigned int)la | ((unsigned int)lb << 16);
  return r;
}
__device__ __forceinline__ v8f wmma_bf16(v16bf a, v16bf b, v8f c) {
  // (neg_a, A, neg_b, B, c_mod, C, reuse_a, reuse_b)
  return __builtin_amdgcn_wmma_f32_16x16x32_bf16(false, a, false, b, (short)0, c, false, false);
}
// A fragment (16x32 bf16): lane<16 -> K {0..7,16..23}; lane>=16 -> K {8..15,24..31}
__device__ __forceinline__ v16bf ld_fragA(const unsigned short* p, int kboff) {
  F8frag f;
  f.lo = *(const u32x4*)(p + kboff);
  f.hi = *(const u32x4*)(p + kboff + 16);
  return __builtin_bit_cast(v16bf, f);
}
// B fragment (32x16 bf16): lane<16 -> K {0..7,16..23}; lane>=16 -> K {8..15,24..31}
__device__ __forceinline__ v16bf ld_fragB(const unsigned short* p, int koff) {
  F8frag f;
  f.lo = *(const u32x4*)(p + koff);
  f.hi = *(const u32x4*)(p + koff + 16);
  return __builtin_bit_cast(v16bf, f);
}

// ---------------- precompute: W -> bf16 hi/lo split ----------------
__global__ __launch_bounds__(256) void hrq_convert_w(const float* __restrict__ W,
                                                     unsigned short* __restrict__ whi,
                                                     unsigned short* __restrict__ wlo) {
  int i = blockIdx.x * 256 + threadIdx.x;   // L*K*D = 1,048,576 elements
  float v = W[i];
  unsigned short h = f2bf(v);
  whi[i] = h;
  wlo[i] = f2bf(v - bf2f(h));
}

// ---------------- precompute: codebook row norms ----------------
__global__ __launch_bounds__(256) void hrq_wnorm(const float* __restrict__ W,
                                                 float* __restrict__ w2,
                                                 float* __restrict__ wn) {
  int r = blockIdx.x * 256 + threadIdx.x;   // 4096 rows
  const f32x4* row = (const f32x4*)(W + (size_t)r * HRQ_D);
  float s = 0.f;
  for (int d4 = 0; d4 < HRQ_D / 4; ++d4) {
    f32x4 v = row[d4];
    s += v.x * v.x + v.y * v.y + v.z * v.z + v.w * v.w;
  }
  w2[r] = s;
  wn[r] = sqrtf(s);
}

// ---------------- per-level kernel: distances via WMMA + argmax/LSE ----------------
__global__ __launch_bounds__(256) void hrq_level(const float* __restrict__ z,
                                                 const unsigned short* __restrict__ whi_all,
                                                 const unsigned short* __restrict__ wlo_all,
                                                 const float* __restrict__ w2_all,
                                                 const float* __restrict__ gsum_all,
                                                 int l,
                                                 int* __restrict__ idx_buf,
                                                 float* __restrict__ kl_acc,
                                                 unsigned int* __restrict__ counts_all,
                                                 float* __restrict__ probs) {
  extern __shared__ char smem[];
  unsigned short* rH = (unsigned short*)smem;        // [64][RSTR] bf16 residual hi
  unsigned short* rL = rH + TOK_WG * RSTR;           // residual lo
  unsigned short* wH = rL + TOK_WG * RSTR;           // [64][RSTR] codebook tile hi
  unsigned short* wL = wH + TOK_WG * RSTR;           // codebook tile lo
  float* w2s  = (float*)(wL + TOK_WG * RSTR);        // [512] all row norms^2 for this level
  float* ssub = w2s + HRQ_K;                         // [64][SSTR] score sub-tile
  float* pm   = ssub + TOK_WG * SSTR;                // [256] partial max
  float* pse  = pm + 256;                            // [256] partial sum-exp
  float* psum = pse + 256;                           // [256] partial sum
  int*   pidx = (int*)(psum + 256);                  // [256] partial argmax

  const int tid = threadIdx.x;
  const int n0  = blockIdx.x * TOK_WG;
  const unsigned short* whi = whi_all + (size_t)l * HRQ_K * HRQ_D;
  const unsigned short* wlo = wlo_all + (size_t)l * HRQ_K * HRQ_D;
  const float* w2g   = w2_all   + l * HRQ_K;
  const float* gsum  = gsum_all + l * HRQ_D;
  unsigned int* counts = counts_all + l * HRQ_K;

  // ---- preload k-block 0 codebook tile into registers (overlaps residual phase) ----
  const u32x4* gH4 = (const u32x4*)whi;   // 32 u32x4 per 256-elem row
  const u32x4* gL4 = (const u32x4*)wlo;
  u32x4 regH[8], regL[8];
#pragma unroll
  for (int t = 0; t < 8; ++t) {
    regH[t] = gH4[tid + t * 256];
    regL[t] = gL4[tid + t * 256];
  }

  // ---- residual = z - gsum, split to bf16 hi/lo in LDS (float4 vectorized) ----
  {
    const f32x4* z4   = (const f32x4*)(z + (size_t)n0 * HRQ_D);
    const f32x4* gs4p = (const f32x4*)gsum;
    const int c4 = tid & 63;            // float4 column (constant across t)
    const f32x4 gs = gs4p[c4];
#pragma unroll
    for (int t = 0; t < 16; ++t) {
      int row = (tid >> 6) + t * 4;
      f32x4 v = z4[row * 64 + c4] - gs;
      HiLo p0 = split2(v.x, v.y);
      HiLo p1 = split2(v.z, v.w);
      u32x2 hi2, lo2;
      hi2.x = p0.hi; hi2.y = p1.hi;
      lo2.x = p0.lo; lo2.y = p1.lo;
      *(u32x2*)(rH + row * RSTR + c4 * 4) = hi2;
      *(u32x2*)(rL + row * RSTR + c4 * 4) = lo2;
    }
    // stage all 512 code norms once
    w2s[tid]       = w2g[tid];
    w2s[tid + 256] = w2g[tid + 256];
  }

  const int wave = tid >> 5, lane = tid & 31;
  const int rb   = wave >> 1;            // row-block 0..3 (16 tokens each)
  const int cb0  = (wave & 1) * 2;       // first of 2 col-blocks
  const int lrow = lane & 15;
  const int kbA  = (lane < 16) ? 0 : 8;  // A-frag K sub-offset
  const int kbB  = (lane < 16) ? 0 : 8;  // B-frag K sub-offset
  const int hi8  = (lane < 16) ? 0 : 8;  // C-frag row offset

  // per-thread online-softmax partial: token = tid&63, col segment = tid>>6
  const int tok = tid & 63, seg = tid >> 6;
  float run_m = -3.402823466e38f, run_se = 0.f, run_sum = 0.f;
  int   run_idx = 0;

  for (int kb = 0; kb < NKB; ++kb) {
    const int k0 = kb * KBLK;
    __syncthreads();  // previous compute/scan fully consumed LDS
    // ---- commit staged codebook tile regs -> LDS (b128 stores) ----
    {
      const int c4s = tid & 31, rw = tid >> 5;
#pragma unroll
      for (int t = 0; t < 8; ++t) {
        int row = rw + t * 8;
        *(u32x4*)(wH + row * RSTR + c4s * 8) = regH[t];
        *(u32x4*)(wL + row * RSTR + c4s * 8) = regL[t];
      }
    }
    // ---- issue next tile's global loads (complete during WMMA below) ----
    if (kb + 1 < NKB) {
      int base = (kb + 1) * KBLK * 32;   // u32x4 offset
#pragma unroll
      for (int t = 0; t < 8; ++t) {
        regH[t] = gH4[base + tid + t * 256];
        regL[t] = gL4[base + tid + t * 256];
      }
    }
    if (kb + 2 < NKB) {  // warm L2 two tiles ahead (global_prefetch_b8)
      __builtin_prefetch(&whi[(size_t)(kb + 2) * KBLK * HRQ_D + (size_t)tid * 64], 0, 1);
      __builtin_prefetch(&wlo[(size_t)(kb + 2) * KBLK * HRQ_D + (size_t)tid * 64], 0, 1);
    }
    __syncthreads();

    // ---- 16x16 WMMA tiles: dot(resid, code) with bf16 hi/lo compensation ----
    v8f acc0 = {0.f,0.f,0.f,0.f,0.f,0.f,0.f,0.f};
    v8f acc1 = {0.f,0.f,0.f,0.f,0.f,0.f,0.f,0.f};
    const unsigned short* aHp = &rH[(rb * 16 + lrow) * RSTR];
    const unsigned short* aLp = &rL[(rb * 16 + lrow) * RSTR];
    const unsigned short* b0H = &wH[((cb0    ) * 16 + lrow) * RSTR];
    const unsigned short* b0L = &wL[((cb0    ) * 16 + lrow) * RSTR];
    const unsigned short* b1H = &wH[((cb0 + 1) * 16 + lrow) * RSTR];
    const unsigned short* b1L = &wL[((cb0 + 1) * 16 + lrow) * RSTR];
#pragma unroll
    for (int dc = 0; dc < HRQ_D; dc += 32) {
      v16bf fAH = ld_fragA(aHp + dc, kbA);
      v16bf fAL = ld_fragA(aLp + dc, kbA);
      v16bf fBH = ld_fragB(b0H + dc, kbB);
      v16bf fBL = ld_fragB(b0L + dc, kbB);
      acc0 = wmma_bf16(fAH, fBH, acc0);
      acc0 = wmma_bf16(fAL, fBH, acc0);
      acc0 = wmma_bf16(fAH, fBL, acc0);
      fBH = ld_fragB(b1H + dc, kbB);
      fBL = ld_fragB(b1L + dc, kbB);
      acc1 = wmma_bf16(fAH, fBH, acc1);
      acc1 = wmma_bf16(fAL, fBH, acc1);
      acc1 = wmma_bf16(fAH, fBL, acc1);
    }
    // ---- s_k = 2*dot - ||w_k||^2 -> LDS sub-tile ----
    {
      int c0 = cb0 * 16 + lrow;
      int c1 = c0 + 16;
      float n0w = w2s[k0 + c0], n1w = w2s[k0 + c1];
#pragma unroll
      for (int i = 0; i < 8; ++i) {
        int r = rb * 16 + i + hi8;
        ssub[r * SSTR + c0] = 2.0f * acc0[i] - n0w;
        ssub[r * SSTR + c1] = 2.0f * acc1[i] - n1w;
      }
    }
    __syncthreads();
    // ---- streaming argmax + online softmax: 4 threads per token, 16 cols each ----
    {
      const float* rowp = &ssub[tok * SSTR + seg * 16];
#pragma unroll
      for (int c = 0; c < 16; ++c) {
        float v = rowp[c];
        run_sum += v;
        if (v > run_m) {
          run_se = run_se * __expf(run_m - v) + 1.0f;
          run_m  = v;
          run_idx = k0 + seg * 16 + c;
        } else {
          run_se += __expf(v - run_m);
        }
      }
    }
  }

  // ---- merge 4 partials per token (LSE merge; argmax tie -> smallest global k) ----
  pm[tid] = run_m; pse[tid] = run_se; psum[tid] = run_sum; pidx[tid] = run_idx;
  __syncthreads();
  if (tid < TOK_WG) {
    float m = pm[tid];
    for (int s = 1; s < 4; ++s) m = fmaxf(m, pm[tid + s * 64]);
    float se = 0.f, sum = 0.f;
    int best = 0x7fffffff;
    for (int s = 0; s < 4; ++s) {
      float ms = pm[tid + s * 64];
      se  += pse[tid + s * 64] * __expf(ms - m);
      sum += psum[tid + s * 64];
      int ix = pidx[tid + s * 64];
      if (ms == m && ix < best) best = ix;
    }
    int n = n0 + tid;
    float lse = m + __logf(se);
    float kl  = -LOGK - sum * (1.0f / (float)HRQ_K) + lse;  // ||r||^2 cancels in log_softmax
    kl_acc[n] += 0.1f * kl;                                 // KL_WEIGHT
    idx_buf[n * HRQ_L + l] = best;
    atomicAdd(&counts[best], 1u);
    probs[(size_t)n * (HRQ_L * HRQ_K) + l * HRQ_K + best] = 1.0f;
  }
}

// ---------------- global residual update: gsum += sum_k count_k * W[l,k,:] ----------------
__global__ __launch_bounds__(256) void hrq_gsum_update(const float* __restrict__ W,
                                                       const unsigned int* __restrict__ counts,
                                                       const float* __restrict__ gsum_in,
                                                       float* __restrict__ gsum_out,
                                                       int l) {
  __shared__ float cnt[HRQ_K];
  int t = threadIdx.x;
  cnt[t]       = (float)counts[l * HRQ_K + t];
  cnt[t + 256] = (float)counts[l * HRQ_K + t + 256];
  __syncthreads();
  float s = gsum_in[t];
  const float* Wl = W + (size_t)l * HRQ_K * HRQ_D;
  for (int k = 0; k < HRQ_K; ++k) {
    float c = cnt[k];
    if (c != 0.0f) s += c * Wl[(size_t)k * HRQ_D + t];
  }
  gsum_out[t] = s;
}

// ---------------- finalize: q_out (transposed), norm loss, loss, indices ----------------
__global__ __launch_bounds__(256) void hrq_finalize(const float* __restrict__ W,
                                                    const float* __restrict__ wn,
                                                    const int* __restrict__ idx_buf,
                                                    const float* __restrict__ kl_acc,
                                                    float* __restrict__ q_out,
                                                    float* __restrict__ idx_out,
                                                    float* __restrict__ loss_out) {
  int n  = blockIdx.x * 256 + threadIdx.x;    // 16384 tokens
  int b  = n >> 10;                           // 1024 tokens per batch
  int hw = n & 1023;
  int id[HRQ_L];
  const f32x4* rows[HRQ_L];
  float up[HRQ_L];
#pragma unroll
  for (int l = 0; l < HRQ_L; ++l) {
    id[l]   = idx_buf[n * HRQ_L + l];
    rows[l] = (const f32x4*)(W + ((size_t)l * HRQ_K + id[l]) * HRQ_D);
    up[l]   = wn[l * HRQ_K + id[l]];
    idx_out[n * HRQ_L + l] = (float)id[l];
  }
  // q_out[b, d, h, w] = sum_l W[l, idx, d]   (stores coalesced across threads at fixed d)
  float* qb = q_out + (size_t)b * HRQ_D * 1024 + hw;
  for (int d4 = 0; d4 < HRQ_D / 4; ++d4) {
    f32x4 s = rows[0][d4] + rows[1][d4] + rows[2][d4] + rows[3][d4]
            + rows[4][d4] + rows[5][d4] + rows[6][d4] + rows[7][d4];
    int d = d4 * 4;
    qb[(size_t)(d    ) * 1024] = s.x;
    qb[(size_t)(d + 1) * 1024] = s.y;
    qb[(size_t)(d + 2) * 1024] = s.z;
    qb[(size_t)(d + 3) * 1024] = s.w;
  }
  float nl = 0.f;
#pragma unroll
  for (int i = 0; i < HRQ_L - 1; ++i) {
    float r = fmaxf(up[i + 1] / up[i] * 4.0f, 1.0f) - 1.0f;  // NORM_LOSS_SCALE
    nl += r * r;
  }
  nl *= (1.0f / (HRQ_L - 1));
  float loss_tok = kl_acc[n] + 0.1f * nl;                    // NORM_LOSS_WEIGHT
  atomicAdd(&loss_out[b], loss_tok * (1.0f / 1024.0f));
}

extern "C" void kernel_launch(void* const* d_in, const int* in_sizes, int n_in,
                              void* d_out, int out_size, void* d_ws, size_t ws_size,
                              hipStream_t stream) {
  (void)in_sizes; (void)n_in; (void)ws_size;
  const float* z = (const float*)d_in[0];   // [16,32,32,256]
  const float* W = (const float*)d_in[1];   // [8,512,256]
  // d_in[2] = epoch (unused in eval path)

  // output layout (concatenated, all float)
  float* out       = (float*)d_out;
  float* q_out     = out;                       // 4,194,304
  float* idx_out   = out + 4194304;             //   131,072
  float* loss_out  = out + 4325376;             //        16
  float* probs_out = out + 4325392;             // 67,108,864

  // workspace layout
  char* ws = (char*)d_ws;
  size_t off = 0;
  float* gsum          = (float*)(ws + off); off += (size_t)(HRQ_L + 1) * HRQ_D * 4;   //  9216 B
  unsigned int* counts = (unsigned int*)(ws + off); off += (size_t)HRQ_L * HRQ_K * 4;  // 16384 B
  float* kl_acc        = (float*)(ws + off); off += (size_t)HRQ_N * 4;                 // 65536 B
  size_t zero_bytes = off;                                                             // zero [gsum|counts|kl_acc]
  float* w2            = (float*)(ws + off); off += (size_t)HRQ_L * HRQ_K * 4;
  float* wn            = (float*)(ws + off); off += (size_t)HRQ_L * HRQ_K * 4;
  int*   idx_buf       = (int*)(ws + off);   off += (size_t)HRQ_N * HRQ_L * 4;
  unsigned short* whi  = (unsigned short*)(ws + off); off += (size_t)HRQ_L * HRQ_K * HRQ_D * 2;
  unsigned short* wlo  = (unsigned short*)(ws + off); off += (size_t)HRQ_L * HRQ_K * HRQ_D * 2;

  // zero: probs (one-hot scatter target), loss (atomic target), gsum level 0, counts, kl_acc
  (void)hipMemsetAsync(d_out, 0, (size_t)out_size * sizeof(float), stream);
  (void)hipMemsetAsync(ws, 0, zero_bytes, stream);

  constexpr size_t SMEM_BYTES =
      (size_t)4 * TOK_WG * RSTR * 2   // bf16 tiles: 135,168
      + (size_t)HRQ_K * 4             // w2s: 2,048
      + (size_t)TOK_WG * SSTR * 4     // ssub: 16,640
      + (size_t)256 * 4 * 4;          // partials: 4,096  => 157,952 B (2 WGs/WGP)
  (void)hipFuncSetAttribute((const void*)hrq_level,
                            hipFuncAttributeMaxDynamicSharedMemorySize,
                            (int)SMEM_BYTES);

  hrq_convert_w<<<HRQ_L * HRQ_K * HRQ_D / 256, 256, 0, stream>>>(W, whi, wlo);
  hrq_wnorm<<<HRQ_L * HRQ_K / 256, 256, 0, stream>>>(W, w2, wn);

  for (int l = 0; l < HRQ_L; ++l) {
    hrq_level<<<HRQ_N / TOK_WG, 256, SMEM_BYTES, stream>>>(
        z, whi, wlo, w2, gsum, l, idx_buf, kl_acc, counts, probs_out);
    if (l + 1 < HRQ_L)
      hrq_gsum_update<<<1, 256, 0, stream>>>(W, counts,
                                             gsum + l * HRQ_D, gsum + (l + 1) * HRQ_D, l);
  }

  hrq_finalize<<<HRQ_N / 256, 256, 0, stream>>>(W, wn, idx_buf, kl_acc,
                                                q_out, idx_out, loss_out);
}
